// Pointnet_SA_6227702580013
// MI455X (gfx1250) — matrix operand order repaired and checked by reading, not compiled, not run on hardware
//
#include <hip/hip_runtime.h>
#include <hip/hip_bf16.h>

// ---------------- problem constants ----------------
#define NB   8
#define NP   4096
#define NS   1024
#define NK   64
#define NC   64
#define R2   0.04f
#define NGROUPS (NB * NS)          // 8192

typedef unsigned short ushort_t;
typedef __attribute__((ext_vector_type(16))) __bf16   v16bf;
typedef __attribute__((ext_vector_type(8)))  float    v8f;
typedef __attribute__((ext_vector_type(4)))  unsigned u32x4;

union FragU { u32x4 q[2]; v16bf v; };

__device__ __forceinline__ ushort_t f2bf(float f) {
    union { float f; unsigned u; } v; v.f = f;
    return (ushort_t)((v.u + 0x7FFFu + ((v.u >> 16) & 1u)) >> 16);   // RNE
}
__device__ __forceinline__ unsigned pack2(float a, float b) {
    return (unsigned)f2bf(a) | ((unsigned)f2bf(b) << 16);
}

// ================= Kernel A: ball query (one wave32 per query) =================
__global__ __launch_bounds__(128) void ball_query_kernel(
    const float* __restrict__ xyz, const int* __restrict__ fps,
    float* __restrict__ new_xyz, int* __restrict__ idx_out)
{
    __shared__ float sd[4][64];
    __shared__ int   si[4][64];

    const int lane = threadIdx.x & 31;
    const int w    = threadIdx.x >> 5;
    const int q    = blockIdx.x * 4 + w;            // 0..8191
    const int b    = q >> 10;

    const float* xb = xyz + (size_t)b * NP * 3;
    const int ci = fps[q];
    const float cx = xb[ci * 3 + 0], cy = xb[ci * 3 + 1], cz = xb[ci * 3 + 2];
    if (lane == 0) {
        new_xyz[(size_t)q * 3 + 0] = cx;
        new_xyz[(size_t)q * 3 + 1] = cy;
        new_xyz[(size_t)q * 3 + 2] = cz;
    }

    int   count     = 0;
    float worstVal  = 3.0e38f;
    int   worstSlot = 0;

    for (int base = 0; base < NP; base += 32) {
        const int n = base + lane;
        const float dx = xb[n * 3 + 0] - cx;
        const float dy = xb[n * 3 + 1] - cy;
        const float dz = xb[n * 3 + 2] - cz;
        const float d2 = dx * dx + dy * dy + dz * dz;
        unsigned mask = (unsigned)__builtin_amdgcn_ballot_w32(d2 < R2);
        while (mask) {                               // uniform loop over candidates
            const int c = __builtin_ctz(mask); mask &= mask - 1u;
            union { float f; int i; } u; u.f = d2;
            u.i = __builtin_amdgcn_readlane(u.i, c); // broadcast d2 from lane c
            const float dc = u.f;
            const int   nc = base + c;
            bool recompute = false;
            if (count < NK) {
                if (lane == 0) { sd[w][count] = dc; si[w][count] = nc; }
                ++count;
                recompute = (count == NK);
            } else if (dc < worstVal) {
                if (lane == 0) { sd[w][worstSlot] = dc; si[w][worstSlot] = nc; }
                recompute = true;
            }
            if (recompute) {                         // wave arg-max over 64 LDS entries
                float v0 = sd[w][lane], v1 = sd[w][lane + 32];
                int   s0 = lane,        s1 = lane + 32;
                if (v1 > v0) { v0 = v1; s0 = s1; }
                #pragma unroll
                for (int off = 16; off > 0; off >>= 1) {
                    const float ov = __shfl_xor(v0, off);
                    const int   os = __shfl_xor(s0, off);
                    if (ov > v0) { v0 = ov; s0 = os; }
                }
                worstVal = v0; worstSlot = s0;       // uniform after full butterfly
            }
        }
    }

    // pad unused slots with INT_MAX, sort indices ascending (odd-even, 64 phases)
    if (lane      >= count) si[w][lane]      = 0x7fffffff;
    if (lane + 32 >= count) si[w][lane + 32] = 0x7fffffff;
    for (int ph = 0; ph < 64; ++ph) {
        const int i = (lane << 1) + (ph & 1);
        if (i + 1 < 64) {
            const int a = si[w][i], bb = si[w][i + 1];
            if (bb < a) { si[w][i] = bb; si[w][i + 1] = a; }
        }
    }
    const int first = si[w][0];                      // min valid index
    const int v0 = (lane      < count) ? si[w][lane]      : first;
    const int v1 = (lane + 32 < count) ? si[w][lane + 32] : first;
    idx_out[(size_t)q * NK + lane]      = v0;
    idx_out[(size_t)q * NK + lane + 32] = v1;
}

// ================= Kernel B: grouped MLP via WMMA bf16 =================
// LDS map (bytes):
//   [0      ,12288) W0 frags (96x64  -> 3x4  tiles of 1KB), rows reordered [points;xyz;pad]
//   [12288  ,28672) W1 frags (64x128 -> 2x8  tiles)
//   [28672  ,94208) W2 frags (128x256-> 4x16 tiles)
//   per wave (4 waves): base = 94208 + wv*24576
//       [base, base+16384)  X0 (64x96 bf16, 12288B used) overlaid later by X2 (64x128 bf16)
//       [base+16384, +8192) X1 (64x64 bf16)
#define SMEM_BYTES (94208 + 4 * 24576)

// A fragment: 16x32 bf16 from row-major LDS (documented 16-bit A layout)
__device__ __forceinline__ v16bf load_a(const ushort_t* X, int strideH, int mBase, int kBase, int lane) {
    const int row  = mBase + (lane & 15);
    const int kOff = kBase + ((lane >> 4) << 3);     // hi lanes start at K+8
    const ushort_t* p = X + row * strideH + kOff;
    FragU f;
    f.q[0] = *(const u32x4*)(p);                     // K .. K+7
    f.q[1] = *(const u32x4*)(p + 16);                // K+16 .. K+23
    return f.v;
}
// B fragment: 32x16 bf16, pre-packed so each lane's 16 halves are contiguous
__device__ __forceinline__ v16bf load_b(const ushort_t* Wf, int frag, int lane) {
    const ushort_t* p = Wf + frag * 512 + lane * 16;
    FragU f;
    f.q[0] = *(const u32x4*)(p);
    f.q[1] = *(const u32x4*)(p + 8);
    return f.v;
}
// packed index for element (k,n) of a KxN weight, ntiles = N/16
__device__ __forceinline__ int bpack_idx(int k, int n, int ntiles) {
    const int frag = (k >> 5) * ntiles + (n >> 4);
    const int ln   = (n & 15) | (((k >> 3) & 1) << 4);
    return frag * 512 + ln * 16 + ((k & 7) | (((k >> 4) & 1) << 3));
}

__global__ __launch_bounds__(128) void grouped_mlp_kernel(
    const float* __restrict__ xyz, const float* __restrict__ points,
    const float* __restrict__ w0, const float* __restrict__ w1, const float* __restrict__ w2,
    const int* __restrict__ idx, const float* __restrict__ new_xyz,
    float* __restrict__ out_pts)
{
    extern __shared__ char smem[];
    ushort_t* W0f = (ushort_t*)(smem);
    ushort_t* W1f = (ushort_t*)(smem + 12288);
    ushort_t* W2f = (ushort_t*)(smem + 28672);
    const int tid  = threadIdx.x;
    const int lane = tid & 31;
    const int wv   = tid >> 5;
    ushort_t* X02 = (ushort_t*)(smem + 94208 + wv * 24576);          // X0 then X2
    ushort_t* X1  = (ushort_t*)(smem + 94208 + wv * 24576 + 16384);

    __builtin_prefetch(w2 + tid * 256, 0, 1);        // global_prefetch_b8 warm-up

    // ---- stage weights (once per block), bf16, fragment-packed ----
    for (int e = tid; e < 96 * 64; e += 128) {       // W0: rows [points(64); xyz(3); 0-pad]
        const int k = e >> 6, n = e & 63;
        float val;
        if (k < 64)      val = w0[(3 + k) * 64 + n];
        else if (k < 67) val = w0[(k - 64) * 64 + n];
        else             val = 0.f;
        W0f[bpack_idx(k, n, 4)] = f2bf(val);
    }
    for (int e = tid; e < 64 * 128; e += 128) {
        const int k = e >> 7, n = e & 127;
        W1f[bpack_idx(k, n, 8)] = f2bf(w1[e]);
    }
    for (int e = tid; e < 128 * 256; e += 128) {
        const int k = e >> 8, n = e & 255;
        W2f[bpack_idx(k, n, 16)] = f2bf(w2[e]);
    }
    __syncthreads();

    const int waveGlobal = blockIdx.x * 4 + wv;      // 2048 waves total
    for (int g = waveGlobal; g < NGROUPS; g += 2048) {
        const int b = g >> 10;
        const int* ig = idx + (size_t)g * NK;
        const float* ncp = new_xyz + (size_t)g * 3;
        const float cx = ncp[0], cy = ncp[1], cz = ncp[2];

        // ---- fill X0: 64 rows x 96 cols bf16 = [points(0..63) | xyz-center(64..66) | 0] ----
        #pragma unroll
        for (int rr = 0; rr < 2; ++rr) {
            const int j = lane + rr * 32;
            const int nidx = ig[j];
            const float* pr = points + ((size_t)b * NP + nidx) * NC;
            unsigned* rowp = (unsigned*)(X02 + j * 96);
            #pragma unroll
            for (int q4 = 0; q4 < 16; ++q4) {
                const float4 v = ((const float4*)pr)[q4];
                rowp[q4 * 2 + 0] = pack2(v.x, v.y);
                rowp[q4 * 2 + 1] = pack2(v.z, v.w);
            }
            const float* xp = xyz + ((size_t)b * NP + nidx) * 3;
            rowp[32] = pack2(xp[0] - cx, xp[1] - cy);
            rowp[33] = pack2(xp[2] - cz, 0.f);
            #pragma unroll
            for (int z = 34; z < 48; ++z) rowp[z] = 0u;
        }
        asm volatile("s_wait_dscnt 0" ::: "memory");

        // ---- layer 1: X0(64x96) @ W0 -> X1(64x64), ReLU ----
        // B fragment shared across the 4 M-tiles (4 live accumulators)
        #pragma unroll
        for (int nt = 0; nt < 4; ++nt) {
            v8f accs[4];
            #pragma unroll
            for (int mt = 0; mt < 4; ++mt) accs[mt] = (v8f)0.0f;
            #pragma unroll
            for (int kt = 0; kt < 3; ++kt) {
                const v16bf bb = load_b(W0f, kt * 4 + nt, lane);
                #pragma unroll
                for (int mt = 0; mt < 4; ++mt) {
                    const v16bf a = load_a(X02, 96, mt * 16, kt * 32, lane);
                    accs[mt] = __builtin_amdgcn_wmma_f32_16x16x32_bf16(
                        false, a, false, bb, (short)0, accs[mt], false, false);
                }
            }
            const int col = nt * 16 + (lane & 15);
            #pragma unroll
            for (int mt = 0; mt < 4; ++mt) {
                const int rb = mt * 16 + ((lane >> 4) << 3);
                #pragma unroll
                for (int r = 0; r < 8; ++r)
                    X1[(rb + r) * 64 + col] = f2bf(fmaxf(accs[mt][r], 0.f));
            }
        }
        // ---- layer 2: X1(64x64) @ W1 -> X2(64x128), ReLU (X2 overlays dead X0) ----
        #pragma unroll
        for (int nt = 0; nt < 8; ++nt) {
            v8f accs[4];
            #pragma unroll
            for (int mt = 0; mt < 4; ++mt) accs[mt] = (v8f)0.0f;
            #pragma unroll
            for (int kt = 0; kt < 2; ++kt) {
                const v16bf bb = load_b(W1f, kt * 8 + nt, lane);
                #pragma unroll
                for (int mt = 0; mt < 4; ++mt) {
                    const v16bf a = load_a(X1, 64, mt * 16, kt * 32, lane);
                    accs[mt] = __builtin_amdgcn_wmma_f32_16x16x32_bf16(
                        false, a, false, bb, (short)0, accs[mt], false, false);
                }
            }
            const int col = nt * 16 + (lane & 15);
            #pragma unroll
            for (int mt = 0; mt < 4; ++mt) {
                const int rb = mt * 16 + ((lane >> 4) << 3);
                #pragma unroll
                for (int r = 0; r < 8; ++r)
                    X02[(rb + r) * 128 + col] = f2bf(fmaxf(accs[mt][r], 0.f));
            }
        }
        // ---- layer 3: X2(64x128) @ W2 -> 64x256, ReLU + column max over 64 rows ----
        for (int nt = 0; nt < 16; ++nt) {
            v8f accs[4];
            #pragma unroll
            for (int mt = 0; mt < 4; ++mt) accs[mt] = (v8f)0.0f;
            #pragma unroll
            for (int kt = 0; kt < 4; ++kt) {
                const v16bf bb = load_b(W2f, kt * 16 + nt, lane);
                #pragma unroll
                for (int mt = 0; mt < 4; ++mt) {
                    const v16bf a = load_a(X02, 128, mt * 16, kt * 32, lane);
                    accs[mt] = __builtin_amdgcn_wmma_f32_16x16x32_bf16(
                        false, a, false, bb, (short)0, accs[mt], false, false);
                }
            }
            float rm = 0.f;                          // ReLU floor folded into max
            #pragma unroll
            for (int mt = 0; mt < 4; ++mt)
                #pragma unroll
                for (int r = 0; r < 8; ++r) rm = fmaxf(rm, accs[mt][r]);
            rm = fmaxf(rm, __shfl_xor(rm, 16));      // lane l <-> l+16 share a column
            if (lane < 16) out_pts[(size_t)g * 256 + nt * 16 + lane] = rm;
        }
    }
}

// ================= launch =================
extern "C" void kernel_launch(void* const* d_in, const int* in_sizes, int n_in,
                              void* d_out, int out_size, void* d_ws, size_t ws_size,
                              hipStream_t stream) {
    const float* xyz    = (const float*)d_in[0];
    const float* points = (const float*)d_in[1];
    const int*   fps    = (const int*)  d_in[2];
    const float* w0     = (const float*)d_in[3];
    const float* w1     = (const float*)d_in[4];
    const float* w2     = (const float*)d_in[5];

    float* out      = (float*)d_out;
    float* new_xyz  = out;                    // 8192*3 floats
    float* new_pts  = out + (size_t)NGROUPS * 3;
    int*   idxbuf   = (int*)d_ws;             // 8192*64 ints = 2 MB scratch

    (void)in_sizes; (void)n_in; (void)out_size; (void)ws_size;

    hipFuncSetAttribute(reinterpret_cast<const void*>(&grouped_mlp_kernel),
                        hipFuncAttributeMaxDynamicSharedMemorySize, SMEM_BYTES);

    ball_query_kernel<<<NGROUPS / 4, 128, 0, stream>>>(xyz, fps, new_xyz, idxbuf);
    grouped_mlp_kernel<<<512, 128, SMEM_BYTES, stream>>>(
        xyz, points, w0, w1, w2, idxbuf, new_xyz, new_pts);
}
